// PackageGCN_18124761989442
// MI455X (gfx1250) — matrix operand order repaired and checked by reading, not compiled, not run on hardware
//
#include <hip/hip_runtime.h>

typedef __attribute__((ext_vector_type(16))) _Float16 v16h;
typedef __attribute__((ext_vector_type(8)))  float    v8f;

#define NN  50000      // nodes
#define NE  800000     // edges
#define NG  128        // graphs
#define CIN 16
#define CH  64         // hidden
#define COUT 2

// ---------------- utility kernels ----------------

__global__ void k_fill(float* __restrict__ p, int n, float v) {
    int i = blockIdx.x * blockDim.x + threadIdx.x;
    if (i < n) p[i] = v;
}

__global__ void k_deg_edges(const int* __restrict__ dst, float* __restrict__ deg, int ne) {
    int i = blockIdx.x * blockDim.x + threadIdx.x;
    if (i < ne) atomicAdd(&deg[dst[i]], 1.0f);
}

__global__ void k_dinv(float* __restrict__ deg, int n) {
    int i = blockIdx.x * blockDim.x + threadIdx.x;
    if (i < n) deg[i] = rsqrtf(deg[i]);   // deg >= 1 (self-loop), always valid
}

// ---------------- weight pre-pack into WMMA B-fragment layout ----------------
// Bp[((kk*4 + tile)*32 + lane)*16 + i] = f16( W[k*CH + tile*16 + (lane&15)] ),
//   k = kk*32 + (i>>3)*16 + (lane>>4)*8 + (i&7)   (zero when k >= K)
// Each lane of the GEMM then loads its whole fragment as 32 contiguous bytes.

__global__ void k_pack_w(const float* __restrict__ W, _Float16* __restrict__ Bp,
                         int K, int total) {
    int t = blockIdx.x * blockDim.x + threadIdx.x;
    if (t >= total) return;
    int i    = t & 15;
    int lane = (t >> 4) & 31;
    int tile = (t >> 9) & 3;
    int kk   = t >> 11;
    int k = kk * 32 + ((i >> 3) << 4) + ((lane >> 4) << 3) + (i & 7);
    int n = tile * 16 + (lane & 15);
    Bp[t] = (k < K) ? (_Float16)W[k * CH + n] : (_Float16)0.0f;
}

// ---------------- WMMA GEMM: Y[N x 64] = X[N x K] @ W[K x 64] ----------------
// K is compile-time: all padding resolves statically, A loads are float4s.
// One wave computes a 16-row block across all four 16-col tiles (A reused in regs).

template<int K>
__global__ void k_gemm_wmma(const float* __restrict__ X, const _Float16* __restrict__ Bp,
                            float* __restrict__ Y, int nRowBlocks) {
    constexpr int KSTEPS = (K + 31) / 32;
    const int wave = threadIdx.x >> 5;
    const int rb   = blockIdx.x * (blockDim.x >> 5) + wave;
    if (rb >= nRowBlocks) return;               // wave-uniform exit: EXEC stays all-1s
    const int lane     = threadIdx.x & 31;
    const int laneM    = lane & 15;             // A row / B col / C col
    const int laneHalf = lane >> 4;
    const int row      = rb * 16 + laneM;

    const v16h* __restrict__ bfrag = (const v16h*)Bp;

    v8f c0 = {}, c1 = {}, c2 = {}, c3 = {};
#pragma unroll
    for (int kk = 0; kk < KSTEPS; ++kk) {
        // ---- A fragment: lane holds row `row`; element i -> k = kk*32 + (i>>3)*16 + laneHalf*8 + (i&7)
        const float* ab = X + (size_t)row * K + kk * 32 + laneHalf * 8;
        v16h a;
        {
            float4 q0 = *(const float4*)(ab);
            float4 q1 = *(const float4*)(ab + 4);
            a[0] = (_Float16)q0.x; a[1] = (_Float16)q0.y;
            a[2] = (_Float16)q0.z; a[3] = (_Float16)q0.w;
            a[4] = (_Float16)q1.x; a[5] = (_Float16)q1.y;
            a[6] = (_Float16)q1.z; a[7] = (_Float16)q1.w;
        }
        if (K - kk * 32 >= 32) {                // compile-time condition
            float4 q2 = *(const float4*)(ab + 16);
            float4 q3 = *(const float4*)(ab + 20);
            a[8]  = (_Float16)q2.x; a[9]  = (_Float16)q2.y;
            a[10] = (_Float16)q2.z; a[11] = (_Float16)q2.w;
            a[12] = (_Float16)q3.x; a[13] = (_Float16)q3.y;
            a[14] = (_Float16)q3.z; a[15] = (_Float16)q3.w;
        } else {
#pragma unroll
            for (int i = 8; i < 16; ++i) a[i] = (_Float16)0.0f;
        }

        // ---- B fragments: one 32-byte vector load per tile from packed weights
        const v16h* bp = bfrag + (size_t)(kk * 4) * 32 + lane;
        v16h b0 = bp[0 * 32];
        c0 = __builtin_amdgcn_wmma_f32_16x16x32_f16(false, a, false, b0, (short)0, c0, false, false);
        v16h b1 = bp[1 * 32];
        c1 = __builtin_amdgcn_wmma_f32_16x16x32_f16(false, a, false, b1, (short)0, c1, false, false);
        v16h b2 = bp[2 * 32];
        c2 = __builtin_amdgcn_wmma_f32_16x16x32_f16(false, a, false, b2, (short)0, c2, false, false);
        v16h b3 = bp[3 * 32];
        c3 = __builtin_amdgcn_wmma_f32_16x16x32_f16(false, a, false, b3, (short)0, c3, false, false);
    }

    float* yb = Y + (size_t)rb * 16 * CH;
#pragma unroll
    for (int v = 0; v < 8; ++v) {
        int m = v + laneHalf * 8;               // C layout: VGPR v -> M = v (+8 for upper lanes)
        yb[m * CH +  0 + laneM] = c0[v];
        yb[m * CH + 16 + laneM] = c1[v];
        yb[m * CH + 32 + laneM] = c2[v];
        yb[m * CH + 48 + laneM] = c3[v];
    }
}

// ---------------- edge scatter: A[d] += H[s] * dinv[s]*dinv[d] ----------------

__global__ void k_scatter(const int* __restrict__ src, const int* __restrict__ dst,
                          const float* __restrict__ dinv, const float* __restrict__ H,
                          float* __restrict__ A, int ne) {
    int t = blockIdx.x * blockDim.x + threadIdx.x;
    int e = t >> 6;
    int c = t & 63;
    if (e >= ne) return;
    int s = src[e], d = dst[e];
    float w = dinv[s] * dinv[d];
    atomicAdd(&A[(size_t)d * CH + c], H[(size_t)s * CH + c] * w);
}

// ---------------- self-loop + bias + ReLU: O = relu(A + H*dinv^2 + b) ----------------

__global__ void k_sbr(const float* __restrict__ A, const float* __restrict__ H,
                      const float* __restrict__ dinv, const float* __restrict__ b,
                      float* __restrict__ O, int n) {
    int t = blockIdx.x * blockDim.x + threadIdx.x;
    int i = t >> 6;
    int c = t & 63;
    if (i >= n) return;
    float di = dinv[i];
    float v  = A[(size_t)i * CH + c] + H[(size_t)i * CH + c] * di * di + b[c];
    O[(size_t)i * CH + c] = v > 0.0f ? v : 0.0f;
}

// ---------------- mean pool accumulation ----------------

__global__ void k_pool(const float* __restrict__ H, const int* __restrict__ batch,
                       float* __restrict__ gsum, float* __restrict__ gcnt, int n) {
    int t = blockIdx.x * blockDim.x + threadIdx.x;
    int i = t >> 6;
    int c = t & 63;
    if (i >= n) return;
    int g = batch[i];
    atomicAdd(&gsum[g * CH + c], H[(size_t)i * CH + c]);
    if (c == 0) atomicAdd(&gcnt[g], 1.0f);
}

// ---------------- head: out[g,o] = mean(g) @ Wc + bc ----------------

__global__ void k_head(const float* __restrict__ gsum, const float* __restrict__ gcnt,
                       const float* __restrict__ Wc, const float* __restrict__ bc,
                       float* __restrict__ out) {
    int t = threadIdx.x;                 // 256 threads = 128 graphs x 2 outputs
    int g = t >> 1, o = t & 1;
    float cnt = gcnt[g];
    cnt = cnt > 1.0f ? cnt : 1.0f;
    float inv = 1.0f / cnt;
    float acc = bc[o];
#pragma unroll 8
    for (int c = 0; c < CH; ++c)
        acc += (gsum[g * CH + c] * inv) * Wc[c * COUT + o];
    out[g * COUT + o] = acc;
}

// ---------------- launch ----------------

extern "C" void kernel_launch(void* const* d_in, const int* in_sizes, int n_in,
                              void* d_out, int out_size, void* d_ws, size_t ws_size,
                              hipStream_t stream) {
    const float* x     = (const float*)d_in[0];
    const int*   ei    = (const int*)d_in[1];     // [2, NE] flattened
    const int*   batch = (const int*)d_in[2];
    const float* W1    = (const float*)d_in[3];
    const float* b1    = (const float*)d_in[4];
    const float* W2    = (const float*)d_in[5];
    const float* b2    = (const float*)d_in[6];
    const float* Wc    = (const float*)d_in[7];
    const float* bc    = (const float*)d_in[8];
    float*       out   = (float*)d_out;

    const int* src = ei;
    const int* dst = ei + NE;

    // workspace layout (float offsets; all 32B-aligned)
    float* ws   = (float*)d_ws;
    float* dinv = ws;                                  // NN floats
    _Float16* bp1 = (_Float16*)(ws + NN);              // 1*4*32*16 = 2048 halves (1024 floats)
    _Float16* bp2 = (_Float16*)(ws + NN + 1024);       // 2*4*32*16 = 4096 halves (2048 floats)
    float* buf0 = ws + NN + 1024 + 2048;               // NN*CH floats
    float* buf1 = buf0 + (size_t)NN * CH;              // NN*CH floats
    float* gsum = buf1 + (size_t)NN * CH;              // NG*CH floats
    float* gcnt = gsum + NG * CH;                      // NG floats

    const int nRowBlocks  = NN / 16;                   // 3125 exactly
    const int gemmGrid    = (nRowBlocks + 7) / 8;      // 8 waves per 256-thread block
    const int featThreads = NN * CH;
    const int featGrid    = (featThreads + 255) / 256;
    const int edgeThreads = NE * 64;
    const int edgeGrid    = (edgeThreads + 255) / 256;

    // degrees with self-loops -> dinv ; pack weights into f16 B fragments
    k_fill<<<(NN + 255) / 256, 256, 0, stream>>>(dinv, NN, 1.0f);
    k_deg_edges<<<(NE + 255) / 256, 256, 0, stream>>>(dst, dinv, NE);
    k_dinv<<<(NN + 255) / 256, 256, 0, stream>>>(dinv, NN);
    k_pack_w<<<(2048 + 255) / 256, 256, 0, stream>>>(W1, bp1, CIN, 2048);
    k_pack_w<<<(4096 + 255) / 256, 256, 0, stream>>>(W2, bp2, CH, 4096);

    // ---- conv1: h1 = x @ W1 ; a1 = scatter(h1) ; h1r = relu(a1 + self + b1) ----
    k_gemm_wmma<CIN><<<gemmGrid, 256, 0, stream>>>(x, bp1, buf0, nRowBlocks);
    k_fill<<<featGrid, 256, 0, stream>>>(buf1, featThreads, 0.0f);
    k_scatter<<<edgeGrid, 256, 0, stream>>>(src, dst, dinv, buf0, buf1, NE);
    k_sbr<<<featGrid, 256, 0, stream>>>(buf1, buf0, dinv, b1, buf0, NN);   // h1r -> buf0

    // ---- conv2: h2 = h1r @ W2 ; a2 = scatter(h2) ; h2r = relu(a2 + self + b2) ----
    k_gemm_wmma<CH><<<gemmGrid, 256, 0, stream>>>(buf0, bp2, buf1, nRowBlocks); // h2 -> buf1
    k_fill<<<featGrid, 256, 0, stream>>>(buf0, featThreads, 0.0f);
    k_scatter<<<edgeGrid, 256, 0, stream>>>(src, dst, dinv, buf1, buf0, NE);
    k_sbr<<<featGrid, 256, 0, stream>>>(buf0, buf1, dinv, b2, buf1, NN);   // h2r -> buf1

    // ---- pool + head ----
    k_fill<<<(NG * CH + NG + 255) / 256, 256, 0, stream>>>(gsum, NG * CH + NG, 0.0f);
    k_pool<<<featGrid, 256, 0, stream>>>(buf1, batch, gsum, gcnt, NN);
    k_head<<<1, 256, 0, stream>>>(gsum, gcnt, Wc, bc, out);
}
